// GroupedQueryAttention_1116691497492
// MI455X (gfx1250) — matrix operand order repaired and checked by reading, not compiled, not run on hardware
//
#include <hip/hip_runtime.h>

// ---------------------------------------------------------------------------
// GQA forward for MI455X (gfx1250). All matmuls via v_wmma_f32_16x16x32_bf16
// (wave32 WMMA, fp32 accumulate). GEMMs are register-blocked 64x64 per wave
// (4x4 WMMA tiles) for a 16-WMMA : 16-load inner step.
// ---------------------------------------------------------------------------

typedef __bf16 bf16_t;
typedef __attribute__((ext_vector_type(8)))  __bf16 v8bf;
typedef __attribute__((ext_vector_type(16))) __bf16 v16bf;
typedef __attribute__((ext_vector_type(8)))  float  v8f;

#define BATCH   2
#define SEQ     2048
#define HID     1024
#define NH      16
#define NKV     4
#define HD      64
#define SCALE   0.125f   // 1/sqrt(64)

__device__ __forceinline__ v16bf pack16(v8bf lo, v8bf hi) {
  v16bf r;
#pragma unroll
  for (int i = 0; i < 8; ++i) { r[i] = lo[i]; r[i + 8] = hi[i]; }
  return r;
}

// A-fragment (16x32 bf16, MxK): lane holds row (lane&15); K elems
// {half*8 .. half*8+7, 16+half*8 .. 16+half*8+7}. rowptr = A + row*ld.
__device__ __forceinline__ v16bf load_a32(const bf16_t* rowptr, int k0, int half) {
  const bf16_t* p = rowptr + k0 + half * 8;
  return pack16(*(const v8bf*)p, *(const v8bf*)(p + 16));
}

// B-fragment (32x16 bf16, KxN) from B^T stored row-major [N,K]: lane holds
// column (lane&15); K elems {half*8 .. half*8+7, 16+half*8 .. 16+half*8+7}. colptr = Bt + col*ld.
__device__ __forceinline__ v16bf load_b32(const bf16_t* colptr, int k0, int half) {
  const bf16_t* p = colptr + k0 + half * 8;
  return pack16(*(const v8bf*)p, *(const v8bf*)(p + 16));
}

__device__ __forceinline__ v8f wmma_bf16(v16bf a, v16bf b, v8f c) {
  return __builtin_amdgcn_wmma_f32_16x16x32_bf16(false, a, false, b,
                                                 (short)0, c, false, false);
}

// ---------------------------------------------------------------------------
// Conversion kernels
// ---------------------------------------------------------------------------
__global__ void cvt_f32_to_bf16(const float* __restrict__ in,
                                bf16_t* __restrict__ out, int n) {
  int i = blockIdx.x * blockDim.x + threadIdx.x;
  if (i < n) out[i] = (bf16_t)in[i];
}

// W is [K,N] row-major fp32 -> Wt is [N,K] row-major bf16
__global__ void transpose_to_bf16(const float* __restrict__ W,
                                  bf16_t* __restrict__ Wt, int K, int N) {
  int i = blockIdx.x * blockDim.x + threadIdx.x;
  if (i < K * N) {
    int nn = i / K, kk = i % K;
    Wt[i] = (bf16_t)W[kk * N + nn];
  }
}

// ---------------------------------------------------------------------------
// Register-blocked 64x64 GEMM core: C[64x64] += A[64xK] * Bt[64xK]^T.
// Returns 16 accumulator tiles acc[i][j]; i = M subtile, j = N subtile.
// ---------------------------------------------------------------------------
__device__ __forceinline__ void gemm64x64(const bf16_t* __restrict__ Abase,
                                          const bf16_t* __restrict__ Btbase,
                                          int K, int ln, int half,
                                          v8f acc[4][4]) {
#pragma unroll
  for (int i = 0; i < 4; ++i)
#pragma unroll
    for (int j = 0; j < 4; ++j) acc[i][j] = (v8f){};

  const bf16_t* arow[4];
  const bf16_t* bcol[4];
#pragma unroll
  for (int i = 0; i < 4; ++i) arow[i] = Abase + (size_t)(i * 16 + ln) * K;
#pragma unroll
  for (int j = 0; j < 4; ++j) bcol[j] = Btbase + (size_t)(j * 16 + ln) * K;

  for (int k0 = 0; k0 < K; k0 += 32) {
    // prefetch next K block (clamped in-range; lowers to global_prefetch_b8)
    int kn = k0 + 32 < K ? k0 + 32 : k0;
    __builtin_prefetch(arow[0] + kn, 0, 1);
    __builtin_prefetch(bcol[0] + kn, 0, 1);

    v16bf a[4], b[4];
#pragma unroll
    for (int i = 0; i < 4; ++i) a[i] = load_a32(arow[i], k0, half);
#pragma unroll
    for (int j = 0; j < 4; ++j) b[j] = load_b32(bcol[j], k0, half);
#pragma unroll
    for (int i = 0; i < 4; ++i)
#pragma unroll
      for (int j = 0; j < 4; ++j)
        acc[i][j] = wmma_bf16(a[i], b[j], acc[i][j]);
  }
}

// ---------------------------------------------------------------------------
// Q projection: [4096,1024] x [1024,1024] -> q_bf[b][h][t][d]
// ---------------------------------------------------------------------------
__global__ __launch_bounds__(256)
void gemm_q_kernel(const bf16_t* __restrict__ xbf,
                   const bf16_t* __restrict__ WqT,
                   bf16_t* __restrict__ qout) {
  const int NT = HID / 64;  // 16
  int wid  = blockIdx.x * (blockDim.x >> 5) + (threadIdx.x >> 5);
  int mt   = wid / NT, nt = wid % NT;
  int lane = threadIdx.x & 31, ln = lane & 15, half = lane >> 4;

  v8f acc[4][4];
  gemm64x64(xbf + (size_t)mt * 64 * HID, WqT + (size_t)nt * 64 * HID,
            HID, ln, half, acc);

#pragma unroll
  for (int j = 0; j < 4; ++j) {
    int n = nt * 64 + j * 16 + ln, h = n >> 6, d = n & 63;
#pragma unroll
    for (int i = 0; i < 4; ++i)
#pragma unroll
      for (int r = 0; r < 8; ++r) {
        int m = mt * 64 + i * 16 + r + half * 8;
        int bb = m >> 11, t = m & (SEQ - 1);
        qout[(((size_t)(bb * NH + h) * SEQ + t) << 6) + d] = (bf16_t)acc[i][j][r];
      }
  }
}

// ---------------------------------------------------------------------------
// KV projection: [4096,1024] x [1024,512] -> k_bf[b][kh][t][d], vT_bf[b][kh][d][t]
// ---------------------------------------------------------------------------
__global__ __launch_bounds__(256)
void gemm_kv_kernel(const bf16_t* __restrict__ xbf,
                    const bf16_t* __restrict__ WkvT,
                    bf16_t* __restrict__ kout,
                    bf16_t* __restrict__ vTout) {
  const int NT = 512 / 64;  // 8
  int wid  = blockIdx.x * (blockDim.x >> 5) + (threadIdx.x >> 5);
  int mt   = wid / NT, nt = wid % NT;
  int lane = threadIdx.x & 31, ln = lane & 15, half = lane >> 4;

  v8f acc[4][4];
  gemm64x64(xbf + (size_t)mt * 64 * HID, WkvT + (size_t)nt * 64 * HID,
            HID, ln, half, acc);

#pragma unroll
  for (int j = 0; j < 4; ++j) {
    int n = nt * 64 + j * 16 + ln;
    if (n < 256) {             // K side: [b][kh][t][d]
      int kh = n >> 6, d = n & 63;
#pragma unroll
      for (int i = 0; i < 4; ++i)
#pragma unroll
        for (int r = 0; r < 8; ++r) {
          int m = mt * 64 + i * 16 + r + half * 8;
          int bb = m >> 11, t = m & (SEQ - 1);
          kout[(((size_t)(bb * NKV + kh) * SEQ + t) << 6) + d] = (bf16_t)acc[i][j][r];
        }
    } else {                   // V side, transposed: [b][kh][d][t]
      int n2 = n - 256, vh = n2 >> 6, d = n2 & 63;
#pragma unroll
      for (int i = 0; i < 4; ++i)
#pragma unroll
        for (int r = 0; r < 8; ++r) {
          int m = mt * 64 + i * 16 + r + half * 8;
          int bb = m >> 11, t = m & (SEQ - 1);
          vTout[((size_t)(bb * NKV + vh) * HD + d) * SEQ + t] = (bf16_t)acc[i][j][r];
        }
    }
  }
}

// ---------------------------------------------------------------------------
// Flash attention, causal, GQA (4 q-heads per kv-head).
// One wave (32 threads) per (b, h, 16-query tile).
// ---------------------------------------------------------------------------
__global__ __launch_bounds__(32)
void attn_kernel(const bf16_t* __restrict__ qbf,
                 const bf16_t* __restrict__ kbf,
                 const bf16_t* __restrict__ vTbf,
                 bf16_t* __restrict__ attn) {
  __shared__ bf16_t psh[16 * 32];   // staged P tile (C-layout -> A-layout)

  const int QT = SEQ / 16;          // 128
  int bid = blockIdx.x;
  int qt = bid % QT;
  int h  = (bid / QT) % NH;
  int b  = bid / (QT * NH);
  int kh = h >> 2;                  // repeat_interleave(GROUPS=4)
  int q0 = qt * 16;

  int lane = threadIdx.x & 31, ln = lane & 15, half = lane >> 4;

  const bf16_t* qbase  = qbf  + (((size_t)(b * NH  + h ) * SEQ + q0) << 6);
  const bf16_t* kbase  = kbf  + (((size_t)(b * NKV + kh) * SEQ) << 6);
  const bf16_t* vTbase = vTbf + ((size_t)(b * NKV + kh) * HD) * SEQ;

  // Q fragments for the whole tile (K=0..31, 32..63)
  const bf16_t* qrow = qbase + (size_t)ln * HD;
  v16bf qa0 = load_a32(qrow, 0,  half);
  v16bf qa1 = load_a32(qrow, 32, half);

  float m_i[8], l_i[8];
  v8f acc[4];
#pragma unroll
  for (int r = 0; r < 8; ++r) { m_i[r] = -1e30f; l_i[r] = 0.0f; }
#pragma unroll
  for (int nt = 0; nt < 4; ++nt) acc[nt] = (v8f){};

  const int kend = q0 + 16;         // keys [0, q0+15] needed
  for (int kb = 0; kb < kend; kb += 32) {
    // scores: two 16x16 tiles over this 32-key block
    v8f s0 = {}, s1 = {};
    const bf16_t* kc0 = kbase + (size_t)(kb + ln) * HD;
    const bf16_t* kc1 = kbase + (size_t)(kb + 16 + ln) * HD;
    s0 = wmma_bf16(qa0, load_b32(kc0, 0,  half), s0);
    s0 = wmma_bf16(qa1, load_b32(kc0, 32, half), s0);
    s1 = wmma_bf16(qa0, load_b32(kc1, 0,  half), s1);
    s1 = wmma_bf16(qa1, load_b32(kc1, 32, half), s1);

    float alpha[8];
#pragma unroll
    for (int r = 0; r < 8; ++r) {
      int qrow_g = q0 + r + half * 8;
      float v0 = s0[r] * SCALE;
      float v1 = s1[r] * SCALE;
      if (kb + ln      > qrow_g) v0 = -1e30f;   // causal mask
      if (kb + 16 + ln > qrow_g) v1 = -1e30f;
      float mx = fmaxf(v0, v1);
#pragma unroll
      for (int off = 1; off < 16; off <<= 1)
        mx = fmaxf(mx, __shfl_xor(mx, off));
      float mnew = fmaxf(m_i[r], mx);
      float a = __expf(m_i[r] - mnew);
      m_i[r] = mnew;
      alpha[r] = a;
      float p0 = __expf(v0 - mnew);
      float p1 = __expf(v1 - mnew);
      float rs = p0 + p1;
#pragma unroll
      for (int off = 1; off < 16; off <<= 1)
        rs += __shfl_xor(rs, off);
      l_i[r] = l_i[r] * a + rs;
      psh[(r + half * 8) * 32 + ln]      = (bf16_t)p0;
      psh[(r + half * 8) * 32 + 16 + ln] = (bf16_t)p1;
    }
    __syncthreads();

    // P(16x32) x V(32x64): A-fragment from LDS, B from transposed V
    v16bf pa = load_a32(psh + ln * 32, 0, half);
#pragma unroll
    for (int nt = 0; nt < 4; ++nt) {
      v8f t = acc[nt];
#pragma unroll
      for (int r = 0; r < 8; ++r) t[r] *= alpha[r];
      const bf16_t* vc = vTbase + (size_t)(nt * 16 + ln) * SEQ + kb;
      acc[nt] = wmma_bf16(pa, load_b32(vc, 0, half), t);
    }
    __syncthreads();
  }

  // normalize and store to attn[b][t][h*64 + d] (bf16, feeds O-projection)
#pragma unroll
  for (int nt = 0; nt < 4; ++nt) {
#pragma unroll
    for (int r = 0; r < 8; ++r) {
      int mrow = q0 + r + half * 8;
      float ov = acc[nt][r] / l_i[r];
      attn[((size_t)(b * SEQ + mrow) * HID) + h * 64 + nt * 16 + ln] = (bf16_t)ov;
    }
  }
}

// ---------------------------------------------------------------------------
// Output projection: attn[4096,1024] x Wo[1024,1024] -> out fp32 (d_out)
// ---------------------------------------------------------------------------
__global__ __launch_bounds__(256)
void gemm_o_kernel(const bf16_t* __restrict__ attn,
                   const bf16_t* __restrict__ WoT,
                   float* __restrict__ out) {
  const int NT = HID / 64;  // 16
  int wid  = blockIdx.x * (blockDim.x >> 5) + (threadIdx.x >> 5);
  int mt   = wid / NT, nt = wid % NT;
  int lane = threadIdx.x & 31, ln = lane & 15, half = lane >> 4;

  v8f acc[4][4];
  gemm64x64(attn + (size_t)mt * 64 * HID, WoT + (size_t)nt * 64 * HID,
            HID, ln, half, acc);

#pragma unroll
  for (int j = 0; j < 4; ++j) {
    int n = nt * 64 + j * 16 + ln;
#pragma unroll
    for (int i = 0; i < 4; ++i)
#pragma unroll
      for (int r = 0; r < 8; ++r) {
        int m = mt * 64 + i * 16 + r + half * 8;
        out[(size_t)m * HID + n] = acc[i][j][r];
      }
  }
}

// ---------------------------------------------------------------------------
// Host-side launch
// ---------------------------------------------------------------------------
extern "C" void kernel_launch(void* const* d_in, const int* in_sizes, int n_in,
                              void* d_out, int out_size, void* d_ws, size_t ws_size,
                              hipStream_t stream) {
  const float* x   = (const float*)d_in[0];  // [2,2048,1024]
  const float* Wq  = (const float*)d_in[1];  // [1024,1024]
  const float* Wkv = (const float*)d_in[2];  // [1024,512]
  const float* Wo  = (const float*)d_in[3];  // [1024,1024]
  float* out = (float*)d_out;                // [2,2048,1024]

  char* w = (char*)d_ws;
  const size_t ROWS = (size_t)BATCH * SEQ;            // 4096
  bf16_t* x_bf   = (bf16_t*)(w + 0);                  //  8 MB
  bf16_t* WqT_bf = (bf16_t*)(w + 8388608);            //  2 MB
  bf16_t* WkvT_bf= (bf16_t*)(w + 10485760);           //  1 MB
  bf16_t* WoT_bf = (bf16_t*)(w + 11534336);           //  2 MB
  bf16_t* q_bf   = (bf16_t*)(w + 13631488);           //  8 MB  [b][h][t][d]
  bf16_t* k_bf   = (bf16_t*)(w + 22020096);           //  2 MB  [b][kh][t][d]
  bf16_t* vT_bf  = (bf16_t*)(w + 24117248);           //  2 MB  [b][kh][d][t]
  bf16_t* attn_bf= (bf16_t*)(w + 26214400);           //  8 MB  [b][t][1024]

  // conversions
  {
    int n = (int)(ROWS * HID);
    cvt_f32_to_bf16<<<(n + 255) / 256, 256, 0, stream>>>(x, x_bf, n);
  }
  transpose_to_bf16<<<(HID * HID + 255) / 256, 256, 0, stream>>>(Wq,  WqT_bf,  HID, HID);
  transpose_to_bf16<<<(HID * 512 + 255) / 256, 256, 0, stream>>>(Wkv, WkvT_bf, HID, 512);
  transpose_to_bf16<<<(HID * HID + 255) / 256, 256, 0, stream>>>(Wo,  WoT_bf,  HID, HID);

  // projections (64x64 per wave => 4x4 WMMA tiles)
  {
    int waves = (int)(ROWS / 64) * (HID / 64);        // 1024 waves
    gemm_q_kernel<<<waves / 8, 256, 0, stream>>>(x_bf, WqT_bf, q_bf);
  }
  {
    int waves = (int)(ROWS / 64) * (512 / 64);        // 512 waves
    gemm_kv_kernel<<<waves / 8, 256, 0, stream>>>(x_bf, WkvT_bf, k_bf, vT_bf);
  }

  // attention: one wave per (b, h, 16-query tile)
  attn_kernel<<<BATCH * NH * (SEQ / 16), 32, 0, stream>>>(q_bf, k_bf, vT_bf, attn_bf);

  // output projection -> fp32 d_out
  {
    int waves = (int)(ROWS / 64) * (HID / 64);        // 1024 waves
    gemm_o_kernel<<<waves / 8, 256, 0, stream>>>(attn_bf, WoT_bf, out);
  }
}
